// GNN_45561013076581
// MI455X (gfx1250) — matrix operand order repaired and checked by reading, not compiled, not run on hardware
//
#include <hip/hip_runtime.h>

// RGCN for MI455X (gfx1250, wave32). GEMMs via v_wmma_f32_16x16x32_f16;
// edge aggregation via L2-resident f32 atomics; fused ReLU+residual+LayerNorm.

#define NN 50000
#define EE 800000
#define DD 128
#define RR 4
#define LL 3

typedef __attribute__((ext_vector_type(16))) _Float16 v16h;
typedef __attribute__((ext_vector_type(8)))  _Float16 v8h;
typedef __attribute__((ext_vector_type(8)))  float    v8f;

// ---------------- workspace layout (bytes, all 256B aligned) ----------------
#define OFF_INV  0u                                  // N*R floats (int cnt first)
#define OFF_AGG  (OFF_INV + (size_t)NN*RR*4)         // N*D f32
#define OFF_HR   (OFF_AGG + (size_t)NN*DD*4)         // N*D f32
#define OFF_XH   (OFF_HR  + (size_t)NN*DD*4)         // N*D f16
#define OFF_WT   (OFF_XH  + (size_t)NN*DD*2)         // 5*D*D f16 (W^T per layer)

// ---------------- small helper kernels ----------------
__global__ void rgcn_zero_i32(int* __restrict__ p, int n) {
    int i = blockIdx.x * blockDim.x + threadIdx.x;
    if (i < n) p[i] = 0;
}

__global__ void rgcn_count(const int* __restrict__ dst, const int* __restrict__ et,
                           int* __restrict__ cnt) {
    int e = blockIdx.x * blockDim.x + threadIdx.x;
    if (e < EE) atomicAdd(&cnt[dst[e] * RR + et[e]], 1);
}

__global__ void rgcn_inv(float* __restrict__ p) {
    int i = blockIdx.x * blockDim.x + threadIdx.x;
    if (i < NN * RR) {
        int c = ((const int*)p)[i];
        if (c < 1) c = 1;
        p[i] = 1.0f / (float)c;
    }
}

// W: [R,D,D] row-major (k-major per matrix), root: [D,D].
// Wt[m][n*D + k] = (f16) src_m[k*D + n]   (m = 0..3 relations, 4 = root)
__global__ void rgcn_prepw(const float* __restrict__ Wl, const float* __restrict__ rootl,
                           _Float16* __restrict__ Wt) {
    int idx = blockIdx.x * blockDim.x + threadIdx.x;
    if (idx >= 5 * DD * DD) return;
    int m   = idx / (DD * DD);
    int rem = idx - m * DD * DD;
    int n   = rem / DD;
    int k   = rem - n * DD;
    const float* srcm = (m < RR) ? (Wl + (size_t)m * DD * DD) : rootl;
    Wt[idx] = (_Float16)srcm[k * DD + n];
}

__global__ void rgcn_cvt_f16(const float* __restrict__ xin, _Float16* __restrict__ Xh) {
    int i = blockIdx.x * blockDim.x + threadIdx.x;
    if (i < NN * DD / 4) {
        float4 v = ((const float4*)xin)[i];
        _Float16* o = Xh + (size_t)i * 4;
        o[0] = (_Float16)v.x; o[1] = (_Float16)v.y;
        o[2] = (_Float16)v.z; o[3] = (_Float16)v.w;
    }
}

// ---------------- WMMA GEMM: Hout[N,D] = Xh[N,D] @ W (+bias) ----------------
// Wt is W transposed: Wt[n*D + k] = W[k][n]. One wave per 16x16 output tile,
// 8 waves per block cover the 8 N-tiles of one 16-row stripe (shared A in L0).
// N % 16 == 0 -> no partial tiles, EXEC all-ones as WMMA requires.
__global__ __launch_bounds__(256)
void rgcn_gemm_f16(const _Float16* __restrict__ Xh, const _Float16* __restrict__ Wt,
                   float* __restrict__ Hout, const float* __restrict__ bias) {
    const int lane  = threadIdx.x;       // 0..31
    const int tileN = threadIdx.y;       // 0..7
    const int tileM = blockIdx.x;        // 0..N/16-1
    const int r     = lane & 15;
    const bool hi   = lane >= 16;

    const int m = tileM * 16 + r;        // A row for this lane
    const int n = tileN * 16 + r;        // B column / C column for this lane

    v8f c = {};
    #pragma unroll
    for (int kk = 0; kk < DD; kk += 32) {
        // A fragment (16x32 f16): lane holds row m, K = kk+(hi?8:0)..+7 and +16..+23
        const _Float16* ap = Xh + (size_t)m * DD + kk + (hi ? 8 : 0);
        v8h a0 = *(const v8h*)(ap);
        v8h a1 = *(const v8h*)(ap + 16);
        v16h a;
        #pragma unroll
        for (int i = 0; i < 8; ++i) { a[i] = a0[i]; a[i + 8] = a1[i]; }
        // B fragment (32x16 f16): lane holds col n, K = kk+(hi?8:0)..+7 and +16..+23
        const _Float16* bp = Wt + (size_t)n * DD + kk + (hi ? 8 : 0);
        v8h b0 = *(const v8h*)(bp);
        v8h b1 = *(const v8h*)(bp + 16);
        v16h b;
        #pragma unroll
        for (int i = 0; i < 8; ++i) { b[i] = b0[i]; b[i + 8] = b1[i]; }
        c = __builtin_amdgcn_wmma_f32_16x16x32_f16(false, a, false, b,
                                                   (short)0, c, false, false);
    }
    const float bv = bias ? bias[n] : 0.0f;
    const int mbase = tileM * 16 + (hi ? 8 : 0);   // C/D: VGPR j -> row mbase+j, col n
    #pragma unroll
    for (int j = 0; j < 8; ++j)
        Hout[(size_t)(mbase + j) * DD + n] = c[j] + bv;
}

// ---------------- edge scatter: agg[dst] += Hr[src] * inv[dst*R+r] ----------------
// One wave per edge; lane handles 4 consecutive floats (float4 gather + 4 atomics).
__global__ __launch_bounds__(256)
void rgcn_scatter(const float* __restrict__ Hr, const int* __restrict__ src,
                  const int* __restrict__ dst, const int* __restrict__ et,
                  const float* __restrict__ inv, float* __restrict__ agg, int rel) {
    const int wave = threadIdx.x >> 5;
    const int lane = threadIdx.x & 31;
    const int e = blockIdx.x * 8 + wave;
    if (e >= EE) return;
    if (et[e] != rel) return;
    const int s = src[e];
    const int d = dst[e];
    const float w = inv[d * RR + rel];
    const float4 h = ((const float4*)(Hr + (size_t)s * DD))[lane];
    float* ap = agg + (size_t)d * DD + lane * 4;
    atomicAdd(ap + 0, h.x * w);
    atomicAdd(ap + 1, h.y * w);
    atomicAdd(ap + 2, h.z * w);
    atomicAdd(ap + 3, h.w * w);
}

// ---------------- finalize: out = LN(relu(agg) + x_res) ----------------
// One wave per node; lane holds 4 floats; wave32 xor-shuffle reductions.
__global__ __launch_bounds__(256)
void rgcn_finalize(const float* __restrict__ agg, const float* __restrict__ xin,
                   const float* __restrict__ gamma, const float* __restrict__ beta,
                   float* __restrict__ xout) {
    const int wave = threadIdx.x >> 5;
    const int lane = threadIdx.x & 31;
    const int node = blockIdx.x * 8 + wave;
    if (node >= NN) return;
    const float4 a  = ((const float4*)(agg + (size_t)node * DD))[lane];
    const float4 x0 = ((const float4*)(xin + (size_t)node * DD))[lane];
    float4 v;
    v.x = fmaxf(a.x, 0.0f) + x0.x;
    v.y = fmaxf(a.y, 0.0f) + x0.y;
    v.z = fmaxf(a.z, 0.0f) + x0.z;
    v.w = fmaxf(a.w, 0.0f) + x0.w;
    float s  = v.x + v.y + v.z + v.w;
    float ss = v.x * v.x + v.y * v.y + v.z * v.z + v.w * v.w;
    #pragma unroll
    for (int off = 16; off > 0; off >>= 1) {
        s  += __shfl_xor(s,  off, 32);
        ss += __shfl_xor(ss, off, 32);
    }
    const float mean = s * (1.0f / DD);
    const float var  = ss * (1.0f / DD) - mean * mean;
    const float rs   = rsqrtf(var + 1e-5f);
    const float4 g = ((const float4*)gamma)[lane];
    const float4 b = ((const float4*)beta)[lane];
    float4 o;
    o.x = (v.x - mean) * rs * g.x + b.x;
    o.y = (v.y - mean) * rs * g.y + b.y;
    o.z = (v.z - mean) * rs * g.z + b.z;
    o.w = (v.w - mean) * rs * g.w + b.w;
    ((float4*)(xout + (size_t)node * DD))[lane] = o;
}

// ---------------- launcher ----------------
extern "C" void kernel_launch(void* const* d_in, const int* in_sizes, int n_in,
                              void* d_out, int out_size, void* d_ws, size_t ws_size,
                              hipStream_t stream) {
    const float* x     = (const float*)d_in[0];   // [N,D]
    const int*   ei    = (const int*)  d_in[1];   // [2,E]
    const int*   et    = (const int*)  d_in[2];   // [E]
    const float* W     = (const float*)d_in[3];   // [L,R,D,D]
    const float* root  = (const float*)d_in[4];   // [L,D,D]
    const float* bias  = (const float*)d_in[5];   // [L,D]
    const float* gamma = (const float*)d_in[6];   // [L,D]
    const float* beta  = (const float*)d_in[7];   // [L,D]
    float* out = (float*)d_out;                   // [N,D]

    char* ws = (char*)d_ws;
    float*    inv = (float*)(ws + OFF_INV);
    float*    agg = (float*)(ws + OFF_AGG);
    float*    Hr  = (float*)(ws + OFF_HR);
    _Float16* Xh  = (_Float16*)(ws + OFF_XH);
    _Float16* Wt  = (_Float16*)(ws + OFF_WT);

    const int* srcI = ei;        // edge_index[0]
    const int* dstI = ei + EE;   // edge_index[1]

    // Per-(dst,rel) mean normalizers: identical for all layers -> compute once.
    rgcn_zero_i32<<<(NN * RR + 255) / 256, 256, 0, stream>>>((int*)inv, NN * RR);
    rgcn_count<<<(EE + 255) / 256, 256, 0, stream>>>(dstI, et, (int*)inv);
    rgcn_inv<<<(NN * RR + 255) / 256, 256, 0, stream>>>(inv);

    const dim3 gblk(32, 8);  // 8 waves: one per 16-column tile of the 128-wide output
    for (int l = 0; l < LL; ++l) {
        const float* xin = (l == 0) ? x : out;   // finalize is per-node -> in-place safe
        rgcn_prepw<<<(5 * DD * DD + 255) / 256, 256, 0, stream>>>(
            W + (size_t)l * RR * DD * DD, root + (size_t)l * DD * DD, Wt);
        rgcn_cvt_f16<<<(NN * DD / 4 + 255) / 256, 256, 0, stream>>>(xin, Xh);
        // agg = x @ root + bias  (initializes agg; no zeroing pass needed)
        rgcn_gemm_f16<<<NN / 16, gblk, 0, stream>>>(Xh, Wt + 4 * DD * DD, agg,
                                                    bias + (size_t)l * DD);
        for (int r = 0; r < RR; ++r) {
            rgcn_gemm_f16<<<NN / 16, gblk, 0, stream>>>(Xh, Wt + (size_t)r * DD * DD,
                                                        Hr, nullptr);
            rgcn_scatter<<<EE / 8, 256, 0, stream>>>(Hr, srcI, dstI, et, inv, agg, r);
        }
        rgcn_finalize<<<NN / 8, 256, 0, stream>>>(agg, xin, gamma + (size_t)l * DD,
                                                  beta + (size_t)l * DD, out);
    }
}
